// TimeGradPredictionNetwork_8658654069149
// MI455X (gfx1250) — matrix operand order repaired and checked by reading, not compiled, not run on hardware
//
#include <hip/hip_runtime.h>
#include <hip/hip_bf16.h>
#include <cmath>

// ---------------------------------------------------------------------------
// Types
// ---------------------------------------------------------------------------
typedef _Float16 v16h __attribute__((ext_vector_type(16)));
typedef _Float16 h8   __attribute__((ext_vector_type(8)));
typedef float    v8f  __attribute__((ext_vector_type(8)));
typedef unsigned int u32x4 __attribute__((ext_vector_type(4)));
typedef int          i32x4 __attribute__((ext_vector_type(4)));
typedef int          i32x8 __attribute__((ext_vector_type(8)));

#define LRELU(x) ((x) >= 0.0f ? (x) : 0.4f * (x))

__device__ __forceinline__ float sigmoidf_(float x) { return 1.0f / (1.0f + expf(-x)); }

// Deterministic counter-based normal sampler (splitmix64 + Box-Muller).
__device__ __forceinline__ float gaussf_(unsigned long long s) {
    s = s * 6364136223846793005ull + 1442695040888963407ull;
    unsigned long long z = s + 0x9E3779B97F4A7C15ull;
    z = (z ^ (z >> 30)) * 0xBF58476D1CE4E5B9ull;
    z = (z ^ (z >> 27)) * 0x94D049BB133111EBull;
    z ^= (z >> 31);
    unsigned int u1 = (unsigned int)z;
    unsigned int u2 = (unsigned int)(z >> 32);
    float f1 = (float)(u1 | 1u) * 2.3283064365386963e-10f;  // (0,1]
    float f2 = (float)u2 * 2.3283064365386963e-10f;
    float r = sqrtf(-2.0f * logf(f1));
    return r * cosf(6.28318530717958647f * f2);
}

// ---------------------------------------------------------------------------
// Small utility kernels
// ---------------------------------------------------------------------------
__global__ void zero_f32(float* p, int n) {
    int i = blockIdx.x * blockDim.x + threadIdx.x;
    if (i < n) p[i] = 0.0f;
}

// Convert f32 [Nsrc x K] row-major -> f16 [Np x Kp], zero padded.
__global__ void cvt_pad_f16(_Float16* dst, const float* src, int Nsrc, int K, int Kp, int total) {
    int i = blockIdx.x * blockDim.x + threadIdx.x;
    if (i >= total) return;
    int n = i / Kp, kk = i % Kp;
    float v = (n < Nsrc && kk < K) ? src[n * K + kk] : 0.0f;
    dst[i] = (_Float16)v;
}

// dst[b*N + j] = bias[j]  (seed gh buffers with b_hh for h==0 at step 0)
__global__ void fill_rows(float* dst, const float* bias, int N, int total) {
    int i = blockIdx.x * blockDim.x + threadIdx.x;
    if (i >= total) return;
    dst[i] = bias[i % N];
}

// Diffusion constants: sr, srm1, plv, c1, c2 at offsets 0/100/200/300/400.
__global__ void diff_consts_k(float* c) {
    int t = threadIdx.x;
    if (t >= 100) return;
    double acp = 1.0, acp_prev = 1.0, beta_t = 0.0, alpha_t = 1.0;
    for (int i = 0; i <= t; ++i) {
        double beta  = 0.0001 + (0.1 - 0.0001) * (double)i / 99.0;
        double alpha = 1.0 - beta;
        acp_prev = acp;
        acp     *= alpha;
        beta_t = beta; alpha_t = alpha;
    }
    double sr   = sqrt(1.0 / acp);
    double srm1 = sqrt(1.0 / acp - 1.0);
    double pv   = beta_t * (1.0 - acp_prev) / (1.0 - acp);
    double plv  = log(fmax(pv, 1e-20));
    double c1   = beta_t * sqrt(acp_prev) / (1.0 - acp);
    double c2   = (1.0 - acp_prev) * sqrt(alpha_t) / (1.0 - acp);
    c[t] = (float)sr; c[100 + t] = (float)srm1; c[200 + t] = (float)plv;
    c[300 + t] = (float)c1; c[400 + t] = (float)c2;
}

// Precompute step vectors for all diffusion timesteps:
// steps[t*64 + blk*8 + ch] = (silu(silu(emb[t]@p1^T+b1)@p2^T+b2)) @ dp_w^T + dp_b
struct EmbArgs {
    const float *emb_table, *p1_w, *p1_b, *p2_w, *p2_b;
    const float *dp_w[8];
    const float *dp_b[8];
    float* steps;
};
__global__ void emb_steps_k(EmbArgs a) {
    __shared__ float e1[64], e2[64];
    int t = blockIdx.x, j = threadIdx.x;
    const float* et = a.emb_table + t * 32;
    float s = a.p1_b[j];
    for (int i = 0; i < 32; ++i) s += et[i] * a.p1_w[j * 32 + i];
    e1[j] = s * sigmoidf_(s);
    __syncthreads();
    s = a.p2_b[j];
    for (int i = 0; i < 64; ++i) s += e1[i] * a.p2_w[j * 64 + i];
    e2[j] = s * sigmoidf_(s);
    __syncthreads();
    int blk = j >> 3, ch = j & 7;
    s = a.dp_b[blk][ch];
    for (int i = 0; i < 64; ++i) s += e2[i] * a.dp_w[blk][ch * 64 + i];
    a.steps[t * 64 + j] = s;
}

// cond_in = [autoreg(225) | control_all[:, k:k+26, :](702)] -> f16 padded to 928
__global__ void build_cond_k(_Float16* condh, const float* autoreg, const float* control,
                             int k, int total) {
    int idx = blockIdx.x * blockDim.x + threadIdx.x;
    if (idx >= total) return;
    int b = idx / 928, j = idx % 928;
    float v = 0.0f;
    if (j < 225) {
        v = autoreg[b * 225 + j];
    } else if (j < 927) {
        int j2 = j - 225;
        v = control[(b * 128 + k + j2 / 27) * 27 + (j2 % 27)];
    }
    condh[idx] = (_Float16)v;
}

// ---------------------------------------------------------------------------
// WMMA f16 GEMM:  D[M,N] = A[M,K] * B[N,K]^T (+ bias[col])
// Block = one 16-row M-tile x 8 N-tiles. The 16xK A-panel is staged into LDS
// once per block via the Tensor Data Mover (TENSORcnt DMA) when available,
// then all 8 waves consume it via ds_load while B streams from global.
// A/B use contiguous K per lane half (k = 16*(lane/16) + s), same order for both; D per ISA.
// ---------------------------------------------------------------------------
__global__ void __launch_bounds__(256) gemm_f16_wmma(const _Float16* __restrict__ A,
                                                     const _Float16* __restrict__ B,
                                                     const float* __restrict__ bias, int biasN,
                                                     float* __restrict__ D,
                                                     int N, int K) {
    __shared__ _Float16 s_A[16 * 928];  // max K = 928

    int waveId = threadIdx.x >> 5;
    int lane   = threadIdx.x & 31;
    int lane16 = lane & 15, laneHi = lane >> 4;
    int m0     = blockIdx.y << 4;
    int nt     = blockIdx.x * 8 + waveId;
    int tilesN = N >> 4;

    // ---- Stage A-panel (16 x K halfs, row-major) into LDS ----
#if __has_builtin(__builtin_amdgcn_tensor_load_to_lds)
    if (waveId == 0) {
        // Build the Tensor DMA descriptor (D#), ISA 8.3/8.4:
        // 2D tensor: data_size=2B, dims = K x 16, tile = K x 16, row stride K.
        unsigned long long ga = (unsigned long long)(A + (size_t)m0 * K);
        typedef __attribute__((address_space(3))) _Float16 lds_h;
        unsigned lds_off = (unsigned)(unsigned long long)(lds_h*)s_A;
        u32x4 g0;
        g0[0] = 1u;                      // count=1, user descriptor
        g0[1] = lds_off;                 // lds_addr (bytes)
        g0[2] = (unsigned)ga;            // global_addr[31:0]
        g0[3] = (unsigned)((ga >> 32) & 0x01FFFFFFull) | 0x80000000u;  // ga[56:32] | type=2
        i32x8 g1;
        g1[0] = (int)0x00010000u;                      // data_size=1 (2 bytes)
        g1[1] = (int)(((unsigned)K & 0xFFFFu) << 16);  // tensor_dim0[15:0]
        g1[2] = (int)(16u << 16);                      // tensor_dim1[15:0] = 16
        g1[3] = (int)(((unsigned)K & 0xFFFFu) << 16);  // tile_dim0 = K
        g1[4] = 16;                                    // tile_dim1 = 16
        g1[5] = K;                                     // tensor_dim0_stride[31:0]
        g1[6] = 0; g1[7] = 0;
        i32x4 gz; gz[0] = 0; gz[1] = 0; gz[2] = 0; gz[3] = 0;
#if __clang_major__ >= 23
        i32x8 gz8; gz8[0]=0; gz8[1]=0; gz8[2]=0; gz8[3]=0; gz8[4]=0; gz8[5]=0; gz8[6]=0; gz8[7]=0;
        __builtin_amdgcn_tensor_load_to_lds(g0, g1, gz, gz, gz8, 0);
#else
        __builtin_amdgcn_tensor_load_to_lds(g0, g1, gz, gz, 0);
#endif
        __builtin_amdgcn_s_wait_tensorcnt(0);
    }
#else
    for (int i = threadIdx.x * 8; i < 16 * K; i += 256 * 8)
        *(h8*)(s_A + i) = *(const h8*)(A + (size_t)m0 * K + i);
#endif
    __syncthreads();

    if (nt < tilesN) {
        int n0 = nt << 4;
        v8f acc = {};
        const _Float16* Abase = s_A + (size_t)lane16 * K + laneHi * 16;        // LDS
        const _Float16* Bbase = B + (size_t)(n0 + lane16) * K + laneHi * 16;   // global
        for (int k0 = 0; k0 < K; k0 += 32) {
            // A 16x32 f16: lanes 0-15 rows with K 0..15, lanes 16-31 rows with K 16..31
            h8 a0 = *(const h8*)(Abase + k0);
            h8 a1 = *(const h8*)(Abase + k0 + 8);
            // B 32x16 f16: lanes 0-15 cols with K 0..15, lanes 16-31 cols with K 16..31
            h8 b0 = *(const h8*)(Bbase + k0);
            h8 b1 = *(const h8*)(Bbase + k0 + 8);
            v16h av = __builtin_shufflevector(a0, a1, 0,1,2,3,4,5,6,7,8,9,10,11,12,13,14,15);
            v16h bv = __builtin_shufflevector(b0, b1, 0,1,2,3,4,5,6,7,8,9,10,11,12,13,14,15);
            acc = __builtin_amdgcn_wmma_f32_16x16x32_f16(false, av, false, bv, (short)0, acc,
                                                         false, false);
        }
        int col  = n0 + lane16;
        int rowb = m0 + laneHi * 8;
        float bb = (bias && col < biasN) ? bias[col] : 0.0f;
        for (int j = 0; j < 8; ++j)
            D[(size_t)(rowb + j) * N + col] = acc[j] + bb;
    }
}

// GRU gate combine: h = (1-z)*n + z*h_prev ; also emits f16 copy for next GEMM.
__global__ void gru_combine(const float* __restrict__ gi, const float* __restrict__ gh,
                            const float* __restrict__ hprev,
                            float* __restrict__ hout, _Float16* __restrict__ houth) {
    int idx = blockIdx.x * blockDim.x + threadIdx.x;
    if (idx >= 64 * 512) return;
    int b = idx >> 9, j = idx & 511;
    const float* gib = gi + b * 1536;
    const float* ghb = gh + b * 1536;
    float r = sigmoidf_(gib[j]        + ghb[j]);
    float z = sigmoidf_(gib[512 + j]  + ghb[512 + j]);
    float n = tanhf(gib[1024 + j] + r * ghb[1024 + j]);
    float hp = hprev ? hprev[idx] : 0.0f;
    float h = (1.0f - z) * n + z * hp;
    hout[idx]  = h;
    houth[idx] = (_Float16)h;
}

// c = lrelu(lrelu(distr @ cu1^T + b1) @ cu2^T + b2)   [64,100] -> [64,45]
__global__ void cond_net_k(const float* __restrict__ distr,
                           const float* __restrict__ cu1w, const float* __restrict__ cu1b,
                           const float* __restrict__ cu2w, const float* __restrict__ cu2b,
                           float* __restrict__ c_all) {
    __shared__ float t1[22];
    int b = blockIdx.x, tid = threadIdx.x;
    const float* db = distr + b * 112;
    if (tid < 22) {
        float s = cu1b[tid];
        for (int i = 0; i < 100; ++i) s += db[i] * cu1w[tid * 100 + i];
        t1[tid] = LRELU(s);
    }
    __syncthreads();
    if (tid < 45) {
        float s = cu2b[tid];
        for (int i = 0; i < 22; ++i) s += t1[i] * cu2w[tid * 22 + i];
        c_all[b * 45 + tid] = LRELU(s);
    }
}

// ---------------------------------------------------------------------------
// Persistent per-batch diffusion sampler: one block per batch element runs the
// full 100-step reverse chain entirely in LDS, then writes the sample and
// shifts the autoregressive window.
// ---------------------------------------------------------------------------
struct DiffArgs {
    const float* dc_w[8]; const float* dc_b[8];
    const float* cp_w[8]; const float* cp_b[8];
    const float* op_w[8]; const float* op_b[8];
    const float *inp_w, *inp_b, *sp_w, *sp_b, *out_w, *out_b, *an_logs, *an_bias;
    const float *c_all, *steps, *consts;
    float *autoreg, *out;
    int k;
};

__global__ void __launch_bounds__(64) diffusion_sample_k(DiffArgs a) {
    __shared__ float s_wdc[8][16][8][3];
    __shared__ float s_bdc[8][16], s_wcp[8][16], s_bcp[8][16];
    __shared__ float s_wop[8][16][8], s_bop[8][16];
    __shared__ float s_spw[8][8][3], s_spb[8], s_ow[8][3];
    __shared__ float s_inw[8], s_inb[8];
    __shared__ float s_c[45], s_x[45];
    __shared__ float s_h[8][49], s_sk[8][49], s_sp[8][47];

    int b = blockIdx.x, tid = threadIdx.x;

    // Stage eps-network weights into LDS once.
    for (int blk = 0; blk < 8; ++blk) {
        for (int i = tid; i < 384; i += 64) (&s_wdc[blk][0][0][0])[i] = a.dc_w[blk][i];
        if (tid < 16) {
            s_bdc[blk][tid] = a.dc_b[blk][tid];
            s_wcp[blk][tid] = a.cp_w[blk][tid];
            s_bcp[blk][tid] = a.cp_b[blk][tid];
            s_bop[blk][tid] = a.op_b[blk][tid];
        }
        for (int i = tid; i < 128; i += 64) (&s_wop[blk][0][0])[i] = a.op_w[blk][i];
    }
    for (int i = tid; i < 192; i += 64) (&s_spw[0][0][0])[i] = a.sp_w[i];
    if (tid < 8) { s_spb[tid] = a.sp_b[tid]; s_inw[tid] = a.inp_w[tid]; s_inb[tid] = a.inp_b[tid]; }
    if (tid < 24) (&s_ow[0][0])[tid] = a.out_w[tid];

    if (tid < 45) {
        s_c[tid] = a.c_all[b * 45 + tid];
        unsigned long long seed =
            ((((unsigned long long)a.k * 1000003ull + 777ull) * 64ull + b) * 64ull + tid);
        s_x[tid] = gaussf_(seed);
    }
    __syncthreads();

    const float isq2 = 0.70710678118654752f;
    const float isq8 = 0.35355339059327376f;

    for (int t = 99; t >= 0; --t) {
        // Input conv: kernel-1 over circularly padded x (len 45 -> 49).
        if (tid < 49) {
            float xv = s_x[(tid + 43) % 45];
            for (int ch = 0; ch < 8; ++ch) {
                float v = s_inw[ch] * xv + s_inb[ch];
                s_h[ch][tid]  = LRELU(v);
                s_sk[ch][tid] = 0.0f;
            }
        }
        __syncthreads();

        for (int blk = 0; blk < 8; ++blk) {
            int d = 1 << (blk & 1);
            float nh[8], nsk[8];
            if (tid < 49) {
                float stp[8];
                for (int ic = 0; ic < 8; ++ic) stp[ic] = a.steps[t * 64 + blk * 8 + ic];
                float cv = s_c[(tid + 43) % 45];
                float y[16];
                for (int oc = 0; oc < 16; ++oc)
                    y[oc] = s_bdc[blk][oc] + s_wcp[blk][oc] * cv + s_bcp[blk][oc];
                int im = tid - d; if (im < 0)   im += 49;
                int ip = tid + d; if (ip >= 49) ip -= 49;
                for (int ic = 0; ic < 8; ++ic) {
                    float hm = s_h[ic][im]  + stp[ic];
                    float h0 = s_h[ic][tid] + stp[ic];
                    float hp = s_h[ic][ip]  + stp[ic];
                    for (int oc = 0; oc < 16; ++oc)
                        y[oc] += s_wdc[blk][oc][ic][0] * hm +
                                 s_wdc[blk][oc][ic][1] * h0 +
                                 s_wdc[blk][oc][ic][2] * hp;
                }
                float g[8];
                for (int c8 = 0; c8 < 8; ++c8) g[c8] = sigmoidf_(y[c8]) * tanhf(y[8 + c8]);
                for (int oc = 0; oc < 16; ++oc) {
                    float s = s_bop[blk][oc];
                    for (int ic = 0; ic < 8; ++ic) s += s_wop[blk][oc][ic] * g[ic];
                    s = LRELU(s);
                    if (oc < 8) nh[oc] = (s_h[oc][tid] + s) * isq2;
                    else        nsk[oc - 8] = s;
                }
            }
            __syncthreads();
            if (tid < 49) {
                for (int ch = 0; ch < 8; ++ch) {
                    s_h[ch][tid]   = nh[ch];
                    s_sk[ch][tid] += nsk[ch];
                }
            }
            __syncthreads();
        }

        // sp conv (valid, kernel 3): 49 -> 47, applied to skips/sqrt(8)
        if (tid < 47) {
            for (int oc = 0; oc < 8; ++oc) {
                float s = s_spb[oc];
                for (int ic = 0; ic < 8; ++ic)
                    for (int kk = 0; kk < 3; ++kk)
                        s += s_spw[oc][ic][kk] * (s_sk[ic][tid + kk] * isq8);
                s_sp[oc][tid] = LRELU(s);
            }
        }
        __syncthreads();

        // out conv (valid, kernel 3): 47 -> 45 ; then reverse-diffusion update.
        float nx = 0.0f;
        if (tid < 45) {
            float e = a.out_b[0];
            for (int ic = 0; ic < 8; ++ic)
                for (int kk = 0; kk < 3; ++kk)
                    e += s_ow[ic][kk] * s_sp[ic][tid + kk];
            float xv   = s_x[tid];
            float sr   = a.consts[t];
            float srm1 = a.consts[100 + t];
            float plv  = a.consts[200 + t];
            float c1   = a.consts[300 + t];
            float c2   = a.consts[400 + t];
            float x0 = sr * xv - srm1 * e;
            x0 = fminf(1.0f, fmaxf(-1.0f, x0));
            float mean = c1 * x0 + c2 * xv;
            float nz = 0.0f;
            if (t > 0) {
                unsigned long long seed =
                    ((((unsigned long long)a.k * 131ull + 1ull + (unsigned)t) * 131071ull + b) * 64ull + tid);
                nz = expf(0.5f * plv) * gaussf_(seed);
            }
            nx = mean + nz;
        }
        __syncthreads();
        if (tid < 45) s_x[tid] = nx;
        __syncthreads();
    }

    if (tid < 45) {
        float samp = s_x[tid] * expf(-a.an_logs[tid]) - a.an_bias[tid];
        a.out[((size_t)b * 108 + 5 + a.k) * 45 + tid] = samp;
        float t0 = a.autoreg[(b * 5 + 1) * 45 + tid];
        float t1v = a.autoreg[(b * 5 + 2) * 45 + tid];
        float t2 = a.autoreg[(b * 5 + 3) * 45 + tid];
        float t3 = a.autoreg[(b * 5 + 4) * 45 + tid];
        a.autoreg[(b * 5 + 0) * 45 + tid] = t0;
        a.autoreg[(b * 5 + 1) * 45 + tid] = t1v;
        a.autoreg[(b * 5 + 2) * 45 + tid] = t2;
        a.autoreg[(b * 5 + 3) * 45 + tid] = t3;
        a.autoreg[(b * 5 + 4) * 45 + tid] = samp;
    }
}

// ---------------------------------------------------------------------------
// Host orchestration
// ---------------------------------------------------------------------------
extern "C" void kernel_launch(void* const* d_in, const int* in_sizes, int n_in,
                              void* d_out, int out_size, void* d_ws, size_t ws_size,
                              hipStream_t stream) {
    (void)in_sizes; (void)n_in; (void)ws_size;

    // setup_inputs() dict order (depth-first flatten)
    const float* control = (const float*)d_in[1];
    const float* w_ih0 = (const float*)d_in[4];
    const float* w_hh0 = (const float*)d_in[5];
    const float* b_ih0 = (const float*)d_in[6];
    const float* b_hh0 = (const float*)d_in[7];
    const float* w_ih1 = (const float*)d_in[8];
    const float* w_hh1 = (const float*)d_in[9];
    const float* b_ih1 = (const float*)d_in[10];
    const float* b_hh1 = (const float*)d_in[11];
    const float* proj_w = (const float*)d_in[12];
    const float* proj_b = (const float*)d_in[13];
    const float* inp_w = (const float*)d_in[14];
    const float* inp_b = (const float*)d_in[15];
    const float* emb_table = (const float*)d_in[16];
    const float* p1_w = (const float*)d_in[17];
    const float* p1_b = (const float*)d_in[18];
    const float* p2_w = (const float*)d_in[19];
    const float* p2_b = (const float*)d_in[20];
    const float* cu1_w = (const float*)d_in[21];
    const float* cu1_b = (const float*)d_in[22];
    const float* cu2_w = (const float*)d_in[23];
    const float* cu2_b = (const float*)d_in[24];
    const float* blk_dc_w[8]; const float* blk_dc_b[8];
    const float* blk_dp_w[8]; const float* blk_dp_b[8];
    const float* blk_cp_w[8]; const float* blk_cp_b[8];
    const float* blk_op_w[8]; const float* blk_op_b[8];
    for (int i = 0; i < 8; ++i) {
        int base = 25 + 8 * i;
        blk_dc_w[i] = (const float*)d_in[base + 0];
        blk_dc_b[i] = (const float*)d_in[base + 1];
        blk_dp_w[i] = (const float*)d_in[base + 2];
        blk_dp_b[i] = (const float*)d_in[base + 3];
        blk_cp_w[i] = (const float*)d_in[base + 4];
        blk_cp_b[i] = (const float*)d_in[base + 5];
        blk_op_w[i] = (const float*)d_in[base + 6];
        blk_op_b[i] = (const float*)d_in[base + 7];
    }
    const float* sp_w = (const float*)d_in[89];
    const float* sp_b = (const float*)d_in[90];
    const float* out_w = (const float*)d_in[91];
    const float* out_b = (const float*)d_in[92];
    const float* an_logs = (const float*)d_in[93];
    const float* an_bias = (const float*)d_in[94];

    float* out = (float*)d_out;  // [64, 108, 45]

    // Workspace carve (256B aligned bump allocator)
    char* p = (char*)d_ws;
    auto alloc = [&](size_t bytes) -> char* {
        char* r = p;
        p += (bytes + 255) & ~(size_t)255;
        return r;
    };
    _Float16* wih0h  = (_Float16*)alloc((size_t)1536 * 928 * 2);
    _Float16* whh0h  = (_Float16*)alloc((size_t)1536 * 512 * 2);
    _Float16* wih1h  = (_Float16*)alloc((size_t)1536 * 512 * 2);
    _Float16* whh1h  = (_Float16*)alloc((size_t)1536 * 512 * 2);
    _Float16* projh  = (_Float16*)alloc((size_t)112 * 512 * 2);
    _Float16* condh  = (_Float16*)alloc((size_t)64 * 928 * 2);
    _Float16* st0h0  = (_Float16*)alloc((size_t)64 * 512 * 2);
    _Float16* st0h1  = (_Float16*)alloc((size_t)64 * 512 * 2);
    _Float16* curh0h = (_Float16*)alloc((size_t)64 * 512 * 2);
    _Float16* curoh  = (_Float16*)alloc((size_t)64 * 512 * 2);
    float* gi      = (float*)alloc((size_t)64 * 1536 * 4);
    float* gh0     = (float*)alloc((size_t)64 * 1536 * 4);
    float* gh1     = (float*)alloc((size_t)64 * 1536 * 4);
    float* st0f0   = (float*)alloc((size_t)64 * 512 * 4);
    float* st0f1   = (float*)alloc((size_t)64 * 512 * 4);
    float* scrh    = (float*)alloc((size_t)64 * 512 * 4);
    float* distr   = (float*)alloc((size_t)64 * 112 * 4);
    float* c_all   = (float*)alloc((size_t)64 * 45 * 4);
    float* steps   = (float*)alloc((size_t)100 * 64 * 4);
    float* consts  = (float*)alloc((size_t)500 * 4);
    float* autoreg = (float*)alloc((size_t)64 * 225 * 4);

    auto zgrid = [](int n) { return (n + 255) / 256; };

    // ---- setup (deterministic each call) ----
    zero_f32<<<zgrid(out_size), 256, 0, stream>>>(out, out_size);
    zero_f32<<<zgrid(64 * 225), 256, 0, stream>>>(autoreg, 64 * 225);

    cvt_pad_f16<<<zgrid(1536 * 928), 256, 0, stream>>>(wih0h, w_ih0, 1536, 927, 928, 1536 * 928);
    cvt_pad_f16<<<zgrid(1536 * 512), 256, 0, stream>>>(whh0h, w_hh0, 1536, 512, 512, 1536 * 512);
    cvt_pad_f16<<<zgrid(1536 * 512), 256, 0, stream>>>(wih1h, w_ih1, 1536, 512, 512, 1536 * 512);
    cvt_pad_f16<<<zgrid(1536 * 512), 256, 0, stream>>>(whh1h, w_hh1, 1536, 512, 512, 1536 * 512);
    cvt_pad_f16<<<zgrid(112 * 512), 256, 0, stream>>>(projh, proj_w, 100, 512, 512, 112 * 512);

    diff_consts_k<<<1, 128, 0, stream>>>(consts);

    EmbArgs ea;
    ea.emb_table = emb_table; ea.p1_w = p1_w; ea.p1_b = p1_b; ea.p2_w = p2_w; ea.p2_b = p2_b;
    for (int i = 0; i < 8; ++i) { ea.dp_w[i] = blk_dp_w[i]; ea.dp_b[i] = blk_dp_b[i]; }
    ea.steps = steps;
    emb_steps_k<<<100, 64, 0, stream>>>(ea);

    // Step 0 uses h==0, so gh == b_hh.
    fill_rows<<<zgrid(64 * 1536), 256, 0, stream>>>(gh0, b_hh0, 1536, 64 * 1536);
    fill_rows<<<zgrid(64 * 1536), 256, 0, stream>>>(gh1, b_hh1, 1536, 64 * 1536);

    auto gemm = [&](const _Float16* A, const _Float16* B, const float* bias, int biasN,
                    float* D, int M, int N, int K) {
        dim3 grid((N / 16 + 7) / 8, M / 16);
        gemm_f16_wmma<<<grid, 256, 0, stream>>>(A, B, bias, biasN, D, N, K);
    };

    DiffArgs da;
    for (int i = 0; i < 8; ++i) {
        da.dc_w[i] = blk_dc_w[i]; da.dc_b[i] = blk_dc_b[i];
        da.cp_w[i] = blk_cp_w[i]; da.cp_b[i] = blk_cp_b[i];
        da.op_w[i] = blk_op_w[i]; da.op_b[i] = blk_op_b[i];
    }
    da.inp_w = inp_w; da.inp_b = inp_b; da.sp_w = sp_w; da.sp_b = sp_b;
    da.out_w = out_w; da.out_b = out_b; da.an_logs = an_logs; da.an_bias = an_bias;
    da.c_all = c_all; da.steps = steps; da.consts = consts;
    da.autoreg = autoreg; da.out = out;

    const int n_iters = 103;  // T_ALL - SEQLEN - N_LOOKAHEAD
    for (int k = 0; k < n_iters; ++k) {
        bool first = (k == 0);

        build_cond_k<<<zgrid(64 * 928), 256, 0, stream>>>(condh, autoreg, control, k, 64 * 928);

        // GRU layer 0
        gemm(condh, wih0h, b_ih0, 1536, gi, 64, 1536, 928);
        gru_combine<<<zgrid(64 * 512), 256, 0, stream>>>(
            gi, gh0, first ? nullptr : st0f0,
            first ? st0f0 : scrh, first ? st0h0 : curh0h);
        const _Float16* h0h = first ? st0h0 : curh0h;

        // GRU layer 1
        gemm(h0h, wih1h, b_ih1, 1536, gi, 64, 1536, 512);
        gru_combine<<<zgrid(64 * 512), 256, 0, stream>>>(
            gi, gh1, first ? nullptr : st0f1,
            first ? st0f1 : scrh, first ? st0h1 : curoh);
        const _Float16* oh = first ? st0h1 : curoh;

        // Projection -> cond features
        gemm(oh, projh, proj_b, 100, distr, 64, 112, 512);
        cond_net_k<<<64, 64, 0, stream>>>(distr, cu1_w, cu1_b, cu2_w, cu2_b, c_all);

        // 100-step diffusion sampler (persistent per-batch blocks)
        da.k = k;
        diffusion_sample_k<<<64, 64, 0, stream>>>(da);

        if (first) {
            // state0 is reused for every later step -> gh becomes constant.
            gemm(st0h0, whh0h, b_hh0, 1536, gh0, 64, 1536, 512);
            gemm(st0h1, whh1h, b_hh1, 1536, gh1, 64, 1536, 512);
        }
    }
}
